// MultiHeadSelfAttention_13700945674801
// MI455X (gfx1250) — matrix operand order repaired and checked by reading, not compiled, not run on hardware
//
#include <hip/hip_runtime.h>

#define EPS 1e-8f

typedef __attribute__((ext_vector_type(16))) __bf16 v16bf;
typedef __attribute__((ext_vector_type(8)))  __bf16 v8bf;
typedef __attribute__((ext_vector_type(8)))  float  v8f;
typedef __attribute__((ext_vector_type(8)))  unsigned short ushort8;
typedef __attribute__((ext_vector_type(4)))  unsigned short ushort4v;
typedef __attribute__((ext_vector_type(4)))  float float4v;

__device__ __forceinline__ unsigned short f2bf(float f) {
    unsigned int u = __float_as_uint(f);
    u += 0x7FFFu + ((u >> 16) & 1u);          // round-to-nearest-even
    return (unsigned short)(u >> 16);
}

__device__ __forceinline__ v8f wmma_bf16(v16bf a, v16bf b, v8f c) {
    return __builtin_amdgcn_wmma_f32_16x16x32_bf16(false, a, false, b, (short)0, c, false, false);
}

// Async DMA: 16 bytes global -> LDS, tracked by ASYNCcnt (no VGPR round-trip,
// decoupled from the DS in-order pipe so WMMA fragment loads don't stall on it).
__device__ __forceinline__ void async_copy16(const unsigned short* g, unsigned short* l) {
    unsigned lds_off = (unsigned)(unsigned long long)l;   // low 32 bits of flat = LDS offset
    asm volatile("global_load_async_to_lds_b128 %0, %1, off"
                 :: "v"(lds_off), "v"(g) : "memory");
}

__device__ __forceinline__ void wait_async0() {
    asm volatile("s_wait_asynccnt 0x0" ::: "memory");
}

__device__ __forceinline__ float blockReduceSum(float v, float* red) {
    red[threadIdx.x] = v; __syncthreads();
    for (int s = 128; s > 0; s >>= 1) {
        if ((int)threadIdx.x < s) red[threadIdx.x] += red[threadIdx.x + s];
        __syncthreads();
    }
    float r = red[0]; __syncthreads();
    return r;
}

// ---------------------------------------------------------------------------
// Prep kernel 1: s[b,c] = w[b,:] . normalize(Ws[c,:]) + bs[c]   (one block / c)
// ---------------------------------------------------------------------------
__global__ __launch_bounds__(256) void k_s(const float* __restrict__ wv,
                                           const float* __restrict__ Ws,
                                           const float* __restrict__ bs,
                                           float* __restrict__ sbuf) {
    const int c = blockIdx.x, t = threadIdx.x;
    __shared__ float row[512];
    __shared__ float red[256];
    float a0 = Ws[c * 512 + t], a1 = Ws[c * 512 + 256 + t];
    float mean = blockReduceSum(a0 + a1, red) * (1.0f / 512.0f);
    a0 -= mean; a1 -= mean;
    row[t] = a0; row[t + 256] = a1;
    float nrm = blockReduceSum(a0 * a0 + a1 * a1, red);
    float g = 1.0f / (sqrtf(nrm) + EPS);
    if (t < 16) {
        const float* wr = wv + t * 512;
        float acc = 0.0f;
        for (int i = 0; i < 512; ++i) acc += wr[i] * row[i];
        sbuf[t * 512 + c] = acc * g + bs[c];
    }
}

// ---------------------------------------------------------------------------
// Prep kernel 2: Wn (bf16 row-normalized Wqkv), acoef[b,o]=rn*IG[o], bcoef
// ---------------------------------------------------------------------------
__global__ __launch_bounds__(256) void k_wn(const float* __restrict__ Wqkv,
                                            const float* __restrict__ sbuf,
                                            const float* __restrict__ IG,
                                            const float* __restrict__ bqkv,
                                            unsigned short* __restrict__ Wn,
                                            float* __restrict__ acoef,
                                            float* __restrict__ bcoef) {
    const int o = blockIdx.x, t = threadIdx.x;
    __shared__ float row[512];
    __shared__ float red[256];
    float a0 = Wqkv[o * 512 + t], a1 = Wqkv[o * 512 + 256 + t];
    float mean = blockReduceSum(a0 + a1, red) * (1.0f / 512.0f);
    a0 -= mean; a1 -= mean;
    float nrm = blockReduceSum(a0 * a0 + a1 * a1, red);
    float g = 1.0f / (sqrtf(nrm) + EPS);
    float w0 = a0 * g, w1 = a1 * g;
    row[t] = w0; row[t + 256] = w1;
    Wn[o * 512 + t] = f2bf(w0);
    Wn[o * 512 + 256 + t] = f2bf(w1);
    __syncthreads();
    if (t < 16) {
        const float* sr = sbuf + t * 512;
        float acc = 0.0f;
        for (int i = 0; i < 512; ++i) { float m = row[i] * sr[i]; acc += m * m; }
        acoef[t * 1536 + o] = rsqrtf(acc + EPS) * IG[o];
    }
    if (t == 16) bcoef[o] = bqkv[o] * IG[o];
}

// ---------------------------------------------------------------------------
// Prep kernel 3: Wpn = normalize(Wproj) in bf16
// ---------------------------------------------------------------------------
__global__ __launch_bounds__(256) void k_wp(const float* __restrict__ Wproj,
                                            unsigned short* __restrict__ Wpn) {
    const int c = blockIdx.x, t = threadIdx.x;
    __shared__ float red[256];
    float a0 = Wproj[c * 512 + t], a1 = Wproj[c * 512 + 256 + t];
    float mean = blockReduceSum(a0 + a1, red) * (1.0f / 512.0f);
    a0 -= mean; a1 -= mean;
    float nrm = blockReduceSum(a0 * a0 + a1 * a1, red);
    float g = 1.0f / (sqrtf(nrm) + EPS);
    Wpn[c * 512 + t] = f2bf(a0 * g);
    Wpn[c * 512 + 256 + t] = f2bf(a1 * g);
}

// ---------------------------------------------------------------------------
// Prep kernel 4: xsT[b][l][c] = s[b,c]*x[b,c,l]  (bf16, transposed via LDS)
// ---------------------------------------------------------------------------
__global__ __launch_bounds__(256) void k_xs(const float* __restrict__ x,
                                            const float* __restrict__ sbuf,
                                            unsigned short* __restrict__ xsT) {
    const int b = blockIdx.z, c0 = blockIdx.y * 32, l0 = blockIdx.x * 32;
    const int t = threadIdx.x;
    __shared__ float tile[32][33];
    __shared__ float ssh[32];
    if (t < 32) ssh[t] = sbuf[b * 512 + c0 + t];
    {
        int ci = t >> 3, lq = (t & 7) * 4;
        float4v v = *(const float4v*)(x + ((size_t)(b * 512 + c0 + ci)) * 1024 + l0 + lq);
        tile[ci][lq] = v[0]; tile[ci][lq + 1] = v[1];
        tile[ci][lq + 2] = v[2]; tile[ci][lq + 3] = v[3];
    }
    __syncthreads();
    {
        int li = t >> 3, cq = (t & 7) * 4;
        ushort4v o;
#pragma unroll
        for (int u = 0; u < 4; ++u) o[u] = f2bf(ssh[cq + u] * tile[cq + u][li]);
        *(ushort4v*)(xsT + ((size_t)b * 1024 + l0 + li) * 512 + c0 + cq) = o;
    }
}

// ---------------------------------------------------------------------------
// Shared WMMA GEMM mainloop: C(64x128) += A(64x512) * Bt(128x512)^T
// A row-major MxK; Bt row-major NxK (both bf16, K stride 512).
// Tiles streamed with GLOBAL_LOAD_ASYNC_TO_LDS double buffering (ASYNCcnt).
// Fully unrolled so buffer parity is static and accumulators stay pinned
// (avoids v_mov rotation + WMMA->VALU hazard NOPs seen with the rolled loop).
// ---------------------------------------------------------------------------
__device__ __forceinline__ void gemm_tile(const unsigned short* __restrict__ Ag,
                                          const unsigned short* __restrict__ Bg,
                                          unsigned short* Asm, unsigned short* Bsm,
                                          v8f* c) {
    const int t = threadIdx.x;
    const int lane = t & 31, wave = t >> 5;
    const int half = lane >> 4, lm = lane & 15;
    const int ra = t >> 2, sa = (t & 3) * 8;   // A: 64 rows x 32
    const int rb = t >> 1, sb = (t & 1) * 16;  // B: 128 rows x 32

    auto issue = [&](int buf, int k0) {
        async_copy16(Ag + ra * 512 + k0 + sa, Asm + buf * 2048 + ra * 32 + sa);
        async_copy16(Bg + rb * 512 + k0 + sb, Bsm + buf * 4096 + rb * 32 + sb);
        async_copy16(Bg + rb * 512 + k0 + sb + 8, Bsm + buf * 4096 + rb * 32 + sb + 8);
    };

    issue(0, 0);
#pragma unroll
    for (int kk = 0; kk < 16; ++kk) {
        wait_async0();          // my async fills for buffer kk done
        __syncthreads();        // everyone's fills done; prior reads retired
        if (kk + 1 < 16) issue((kk + 1) & 1, (kk + 1) * 32);
        const unsigned short* Ab = Asm + (kk & 1) * 2048;
        const unsigned short* Bb = Bsm + (kk & 1) * 4096;
        v16bf bf = *(const v16bf*)(Bb + (wave * 16 + lm) * 32 + half * 16);
#pragma unroll
        for (int i = 0; i < 4; ++i) {
            const unsigned short* ar = Ab + (i * 16 + lm) * 32 + half * 16;
            v8bf lo = *(const v8bf*)ar;
            v8bf hi = *(const v8bf*)(ar + 8);
            v16bf af = __builtin_shufflevector(lo, hi, 0, 1, 2, 3, 4, 5, 6, 7,
                                               8, 9, 10, 11, 12, 13, 14, 15);
            c[i] = wmma_bf16(af, bf, c[i]);
        }
    }
}

// ---------------------------------------------------------------------------
// GEMM1: qkv = (acoef * (Wn @ xs) + bcoef); q,k L2-normalized over d in-epilogue
// Grid: (nblk=8, mblk=24, b=16). mblk picks one head-section (q/k/v, head).
// Outputs: qT,kT as (b,h,l,d); v as (b,h,d,m)   -- all bf16.
// ---------------------------------------------------------------------------
__global__ __launch_bounds__(256) void k_qkv(const unsigned short* __restrict__ Wn,
                                             const unsigned short* __restrict__ xsT,
                                             const float* __restrict__ acoef,
                                             const float* __restrict__ bcoef,
                                             unsigned short* __restrict__ qT,
                                             unsigned short* __restrict__ kT,
                                             unsigned short* __restrict__ vbuf) {
    __shared__ __align__(32) unsigned short Asm[2 * 2048];
    __shared__ __align__(32) unsigned short Bsm[2 * 4096];
    const int b = blockIdx.z, mblk = blockIdx.y, nblk = blockIdx.x;
    const unsigned short* Ag = Wn + mblk * 64 * 512;
    const unsigned short* Bg = xsT + (size_t)b * 1024 * 512 + nblk * 128 * 512;
    v8f c[4] = {};
    gemm_tile(Ag, Bg, Asm, Bsm, c);

    const int lane = threadIdx.x & 31, wave = threadIdx.x >> 5;
    const int half = lane >> 4, lm = lane & 15;
    const int sec = mblk >> 3, head = mblk & 7;
    const int l = nblk * 128 + wave * 16 + lm;

    float vals[4][8];
#pragma unroll
    for (int i = 0; i < 4; ++i)
#pragma unroll
        for (int j = 0; j < 8; ++j) {
            int o = mblk * 64 + i * 16 + half * 8 + j;
            vals[i][j] = acoef[b * 1536 + o] * c[i][j] + bcoef[o];
        }

    if (sec < 2) {  // q or k : normalize over d=64 per (b,h,l)
        float ss = 0.0f;
#pragma unroll
        for (int i = 0; i < 4; ++i)
#pragma unroll
            for (int j = 0; j < 8; ++j) ss += vals[i][j] * vals[i][j];
        ss += __shfl_xor(ss, 16, 32);
        float inv = rsqrtf(ss + EPS);
        unsigned short* dst = (sec == 0 ? qT : kT) + ((size_t)((b * 8 + head) * 1024 + l)) * 64;
#pragma unroll
        for (int i = 0; i < 4; ++i) {
            ushort8 pk;
#pragma unroll
            for (int j = 0; j < 8; ++j) pk[j] = f2bf(vals[i][j] * inv);
            *(ushort8*)(dst + i * 16 + half * 8) = pk;
        }
    } else {        // v : store (b,h,d,m) layout
        unsigned short* dst = vbuf + (size_t)(b * 8 + head) * 64 * 1024;
#pragma unroll
        for (int i = 0; i < 4; ++i)
#pragma unroll
            for (int j = 0; j < 8; ++j)
                dst[(size_t)(i * 16 + half * 8 + j) * 1024 + l] = f2bf(vals[i][j]);
    }
}

// ---------------------------------------------------------------------------
// Flash attention: per (b,h,lblk) WG; each wave owns 16 queries.
// K (32x64) and V (64x32) m-tiles are async-DMA'd into LDS (double buffered)
// and shared by all 8 waves. S = Q.K^T*8 via WMMA, online softmax,
// O += P.V^T via WMMA (P converted C-frag -> A-frag through per-wave LDS).
// m-loop manually unrolled x2 so the buffer parity is static.
// Writes obufT as (b, l, c) bf16 for the output-projection GEMM.
// ---------------------------------------------------------------------------
__global__ __launch_bounds__(256) void k_attn(const unsigned short* __restrict__ qT,
                                              const unsigned short* __restrict__ kT,
                                              const unsigned short* __restrict__ vbuf,
                                              unsigned short* __restrict__ obufT) {
    __shared__ __align__(32) unsigned short Ksm[2 * 2048];   // [buf][m(32)][d(64)]
    __shared__ __align__(32) unsigned short Vsm[2 * 2048];   // [buf][d(64)][m(32)]
    __shared__ __align__(32) unsigned short Plds[8 * 16 * 32];
    const int b = blockIdx.z, h = blockIdx.y, lblk = blockIdx.x;
    const int t = threadIdx.x, lane = t & 31, wave = t >> 5;
    const int half = lane >> 4, lm = lane & 15;

    const unsigned short* qbase = qT + ((size_t)((b * 8 + h) * 1024 + lblk * 128 + wave * 16)) * 64;
    const unsigned short* kbase = kT + (size_t)(b * 8 + h) * 1024 * 64;
    const unsigned short* vbase = vbuf + (size_t)(b * 8 + h) * 64 * 1024;

    const int rk = t >> 3, sk = (t & 7) * 8;   // K tile: 32 rows x 64
    const int rv = t >> 2, sv = (t & 3) * 8;   // V tile: 64 rows x 32
    auto issueKV = [&](int buf, int m0) {
        async_copy16(kbase + (m0 + rk) * 64 + sk, Ksm + buf * 2048 + rk * 64 + sk);
        async_copy16(vbase + rv * 1024 + m0 + sv, Vsm + buf * 2048 + rv * 32 + sv);
    };

    v16bf qa[2];
#pragma unroll
    for (int tq = 0; tq < 2; ++tq) {
        const unsigned short* qp = qbase + lm * 64 + tq * 32 + half * 16;
        v8bf lo = *(const v8bf*)qp;
        v8bf hi = *(const v8bf*)(qp + 8);
        qa[tq] = __builtin_shufflevector(lo, hi, 0, 1, 2, 3, 4, 5, 6, 7,
                                         8, 9, 10, 11, 12, 13, 14, 15);
    }

    float mrow[8], lrow[8];
#pragma unroll
    for (int j = 0; j < 8; ++j) { mrow[j] = -1e30f; lrow[j] = 0.0f; }
    v8f oc[4] = {};
    unsigned short* Pw = Plds + wave * 512;

    auto attn_step = [&](const unsigned short* Kb, const unsigned short* Vb,
                         int m0, int nbuf, bool doIssue) {
        wait_async0();
        __syncthreads();
        if (doIssue) issueKV(nbuf, m0 + 32);

        v8f sf[2];
#pragma unroll
        for (int t2 = 0; t2 < 2; ++t2) {
            const unsigned short* kp = Kb + (t2 * 16 + lm) * 64;
            v16bf kb0 = *(const v16bf*)(kp + half * 16);
            v16bf kb1 = *(const v16bf*)(kp + 32 + half * 16);
            v8f z = {};
            v8f s = wmma_bf16(qa[0], kb0, z);
            sf[t2] = wmma_bf16(qa[1], kb1, s);
        }
#pragma unroll
        for (int j = 0; j < 8; ++j) {
            float s0 = sf[0][j] * 8.0f;  // scale = sqrt(d) = 8
            float s1 = sf[1][j] * 8.0f;
            float mx = fmaxf(s0, s1);
#pragma unroll
            for (int d = 1; d < 16; d <<= 1) mx = fmaxf(mx, __shfl_xor(mx, d, 32));
            float newm = fmaxf(mrow[j], mx);
            float sc = __expf(mrow[j] - newm);
            float p0 = __expf(s0 - newm);
            float p1 = __expf(s1 - newm);
            float rs = p0 + p1;
#pragma unroll
            for (int d = 1; d < 16; d <<= 1) rs += __shfl_xor(rs, d, 32);
            lrow[j] = lrow[j] * sc + rs;
            mrow[j] = newm;
#pragma unroll
            for (int f = 0; f < 4; ++f) oc[f][j] *= sc;
            int r = half * 8 + j;
            Pw[r * 32 + lm] = f2bf(p0);
            Pw[r * 32 + 16 + lm] = f2bf(p1);
        }
        const unsigned short* pp = Pw + lm * 32 + half * 16;
        v8bf plo = *(const v8bf*)pp;
        v8bf phi = *(const v8bf*)(pp + 8);
        v16bf pa = __builtin_shufflevector(plo, phi, 0, 1, 2, 3, 4, 5, 6, 7,
                                           8, 9, 10, 11, 12, 13, 14, 15);
#pragma unroll
        for (int f = 0; f < 4; ++f) {
            v16bf vb = *(const v16bf*)(Vb + (f * 16 + lm) * 32 + half * 16);
            oc[f] = wmma_bf16(pa, vb, oc[f]);
        }
    };

    issueKV(0, 0);
    for (int it = 0; it < 32; it += 2) {
        attn_step(Ksm, Vsm, it * 32, 1, true);
        attn_step(Ksm + 2048, Vsm + 2048, (it + 1) * 32, 0, it + 2 < 32);
    }

#pragma unroll
    for (int j = 0; j < 8; ++j) {
        float inv = 1.0f / lrow[j];
        int l = lblk * 128 + wave * 16 + half * 8 + j;
#pragma unroll
        for (int f = 0; f < 4; ++f) {
            int cch = h * 64 + f * 16 + lm;
            obufT[((size_t)b * 1024 + l) * 512 + cch] = f2bf(oc[f][j] * inv);
        }
    }
}

// ---------------------------------------------------------------------------
// GEMM2: out = ResidualGain * (Wpn @ o) + x
// Grid: (nblk=8, mblk=8, b=16)
// ---------------------------------------------------------------------------
__global__ __launch_bounds__(256) void k_out(const unsigned short* __restrict__ Wpn,
                                             const unsigned short* __restrict__ obufT,
                                             const float* __restrict__ RG,
                                             const float* __restrict__ x,
                                             float* __restrict__ out) {
    __shared__ __align__(32) unsigned short Asm[2 * 2048];
    __shared__ __align__(32) unsigned short Bsm[2 * 4096];
    const int b = blockIdx.z, mblk = blockIdx.y, nblk = blockIdx.x;
    const unsigned short* Ag = Wpn + mblk * 64 * 512;
    const unsigned short* Bg = obufT + (size_t)b * 1024 * 512 + nblk * 128 * 512;
    v8f c[4] = {};
    gemm_tile(Ag, Bg, Asm, Bsm, c);

    const int lane = threadIdx.x & 31, wave = threadIdx.x >> 5;
    const int half = lane >> 4, lm = lane & 15;
    const int l = nblk * 128 + wave * 16 + lm;
#pragma unroll
    for (int i = 0; i < 4; ++i)
#pragma unroll
        for (int j = 0; j < 8; ++j) {
            int cg = mblk * 64 + i * 16 + half * 8 + j;
            size_t idx = ((size_t)b * 512 + cg) * 1024 + l;
            out[idx] = RG[cg] * c[i][j] + x[idx];
        }
}

// ---------------------------------------------------------------------------
extern "C" void kernel_launch(void* const* d_in, const int* in_sizes, int n_in,
                              void* d_out, int out_size, void* d_ws, size_t ws_size,
                              hipStream_t stream) {
    const float* x     = (const float*)d_in[0];
    const float* wv    = (const float*)d_in[1];
    const float* IG    = (const float*)d_in[2];
    const float* RG    = (const float*)d_in[3];
    const float* Wqkv  = (const float*)d_in[4];
    const float* bqkv  = (const float*)d_in[5];
    const float* Ws    = (const float*)d_in[6];
    const float* bs    = (const float*)d_in[7];
    const float* Wproj = (const float*)d_in[8];
    float* out = (float*)d_out;
    (void)in_sizes; (void)n_in; (void)out_size; (void)ws_size;

    char* ws = (char*)d_ws;
    float* sbuf           = (float*)(ws + 0);               //  32768 B
    float* acoef          = (float*)(ws + 32768);           //  98304 B
    float* bcoef          = (float*)(ws + 131072);          //   6144 B
    unsigned short* Wn    = (unsigned short*)(ws + 137216);   // 1.5 MB
    unsigned short* Wpn   = (unsigned short*)(ws + 1710080);  // 0.5 MB
    unsigned short* xsT   = (unsigned short*)(ws + 2234368);  // 16 MB
    unsigned short* qT    = (unsigned short*)(ws + 19011584); // 16 MB
    unsigned short* kT    = (unsigned short*)(ws + 35788800); // 16 MB
    unsigned short* vbuf  = (unsigned short*)(ws + 52566016); // 16 MB
    unsigned short* obufT = (unsigned short*)(ws + 69343232); // 16 MB

    k_s  <<<512, 256, 0, stream>>>(wv, Ws, bs, sbuf);
    k_wn <<<1536, 256, 0, stream>>>(Wqkv, sbuf, IG, bqkv, Wn, acoef, bcoef);
    k_wp <<<512, 256, 0, stream>>>(Wproj, Wpn);
    k_xs <<<dim3(32, 16, 16), 256, 0, stream>>>(x, sbuf, xsT);
    k_qkv<<<dim3(8, 24, 16), 256, 0, stream>>>(Wn, xsT, acoef, bcoef, qT, kT, vbuf);
    k_attn<<<dim3(8, 8, 16), 256, 0, stream>>>(qT, kT, vbuf, obufT);
    k_out<<<dim3(8, 8, 16), 256, 0, stream>>>(Wpn, obufT, RG, x, out);
}
